// MHA_77867757076838
// MI455X (gfx1250) — matrix operand order repaired and checked by reading, not compiled, not run on hardware
//
#include <hip/hip_runtime.h>
#include <hip/hip_bf16.h>
#include <stdint.h>

typedef __attribute__((ext_vector_type(16))) __bf16 v16bf;
typedef __attribute__((ext_vector_type(8)))  float  v8f;

#define BB 2
#define SS 2048
#define EE 2048
#define HH 32
#define DD 64
#define NEG_MASK -10000.0f

union ABFrag { uint4 q[2]; v16bf v; };

__device__ __forceinline__ uint16_t f2bf(float f) {
  uint32_t u = __float_as_uint(f);
  uint32_t r = (u + 0x7FFFu + ((u >> 16) & 1u)) >> 16;
  return (uint16_t)r;
}

// low 32 bits of a generic LDS pointer == group-segment byte offset
__device__ __forceinline__ uint32_t lds_off32(const void* p) {
  return (uint32_t)(uintptr_t)p;
}

// CDNA5 async global->LDS copy, 16B per lane, tracked by ASYNCcnt
__device__ __forceinline__ void async_b128(uint32_t lds_off, const void* gptr) {
  asm volatile("global_load_async_to_lds_b128 %0, %1, off"
               :: "v"(lds_off), "v"((uint64_t)(uintptr_t)gptr)
               : "memory");
}

// ---------------- stage 0: f32 -> bf16 convert / transpose ----------------
__global__ void cvt_bf16(const float* __restrict__ in, uint16_t* __restrict__ out, int n) {
  int i = blockIdx.x * blockDim.x + threadIdx.x;
  if (i < n) out[i] = f2bf(in[i]);
}

__global__ void cvt_transpose_bf16(const float* __restrict__ in, uint16_t* __restrict__ out,
                                   int rows, int cols) {
  int i = blockIdx.x * blockDim.x + threadIdx.x;
  if (i < rows * cols) {
    int r = i / cols, c = i - r * cols;
    out[(size_t)c * rows + r] = f2bf(in[i]);
  }
}

// ---------------- LDS-tiled WMMA GEMM: C[M,N] = A[M,K] * Bt[N,K]^T + bias ----------------
// Macro tile 128x128, BK=32, 8 waves (4 along M x 2 along N), each wave 2x4 tiles of 16x16.
// Global->LDS staging via async-to-LDS (ASYNCcnt), double buffered.
// LDS row stride 40 bf16 (80B): keeps 16B alignment for b128 DS ops, spreads banks.
#define BKP 40
__global__ __launch_bounds__(256) void gemm_bf16_lds(
    const uint16_t* __restrict__ A, const uint16_t* __restrict__ Bt,
    const float* __restrict__ bias, float* __restrict__ C,
    int M, int N, int K)
{
  __shared__ uint16_t As[2][128][BKP];
  __shared__ uint16_t Bs[2][128][BKP];

  const int tid  = threadIdx.x;
  const int lane = tid & 31;
  const int wave = tid >> 5;
  const int g    = lane >> 4;
  const int ln   = lane & 15;
  const int wm   = wave >> 1;            // 0..3 (M)
  const int wn   = wave & 1;             // 0..1 (N)
  const int m0   = blockIdx.y * 128;
  const int n0   = blockIdx.x * 128;

  // copy assignment: thread owns (row = tid>>1, 16-col half = tid&1) of each 128x32 tile
  const int crow  = tid >> 1;
  const int chalf = tid & 1;
  const uint16_t* agp = A  + (size_t)(m0 + crow) * K + chalf * 16;
  const uint16_t* bgp = Bt + (size_t)(n0 + crow) * K + chalf * 16;
  uint32_t aoff[2], boff[2];
  aoff[0] = lds_off32(&As[0][crow][chalf * 16]);
  aoff[1] = lds_off32(&As[1][crow][chalf * 16]);
  boff[0] = lds_off32(&Bs[0][crow][chalf * 16]);
  boff[1] = lds_off32(&Bs[1][crow][chalf * 16]);

  v8f acc[2][4] = {};

  const int nk = K / 32;
  // prologue: fill buffer 0
  async_b128(aoff[0],      agp);
  async_b128(aoff[0] + 16, agp + 8);
  async_b128(boff[0],      bgp);
  async_b128(boff[0] + 16, bgp + 8);

  for (int i = 0; i < nk; ++i) {
    asm volatile("s_wait_asynccnt 0x0" ::: "memory");
    __syncthreads();                         // buffer i&1 ready for all waves

    if (i + 1 < nk) {                        // prefetch next K-slice into other buffer
      const int buf = (i + 1) & 1;
      const int kk  = (i + 1) * 32;
      async_b128(aoff[buf],      agp + kk);
      async_b128(aoff[buf] + 16, agp + kk + 8);
      async_b128(boff[buf],      bgp + kk);
      async_b128(boff[buf] + 16, bgp + kk + 8);
    }

    const int buf = i & 1;
    ABFrag af[2], bf[4];
#pragma unroll
    for (int tm = 0; tm < 2; ++tm) {
      const uint16_t* ap = &As[buf][wm * 32 + tm * 16 + ln][0];
      af[tm].q[0] = *(const uint4*)(ap + 8 * g);
      af[tm].q[1] = *(const uint4*)(ap + 16 + 8 * g);
    }
#pragma unroll
    for (int tn = 0; tn < 4; ++tn) {
      const uint16_t* bp = &Bs[buf][wn * 64 + tn * 16 + ln][0];
      bf[tn].q[0] = *(const uint4*)(bp + 8 * g);
      bf[tn].q[1] = *(const uint4*)(bp + 16 + 8 * g);
    }
#pragma unroll
    for (int tm = 0; tm < 2; ++tm)
#pragma unroll
      for (int tn = 0; tn < 4; ++tn)
        acc[tm][tn] = __builtin_amdgcn_wmma_f32_16x16x32_bf16(
            false, af[tm].v, false, bf[tn].v, (short)0, acc[tm][tn], false, false);

    __syncthreads();                         // all waves done reading before refill
  }

  // epilogue: bias + store f32
  const int row0 = m0 + wm * 32;
  const int col0 = n0 + wn * 64;
#pragma unroll
  for (int tm = 0; tm < 2; ++tm) {
#pragma unroll
    for (int tn = 0; tn < 4; ++tn) {
      const float bv = bias[col0 + tn * 16 + ln];
      float* cp = C + (size_t)(row0 + tm * 16 + 8 * g) * N + col0 + tn * 16 + ln;
#pragma unroll
      for (int r = 0; r < 8; ++r)
        cp[(size_t)r * N] = acc[tm][tn][r] + bv;
    }
  }
}

// ---------------- stage 2: RoPE + pack Q,K row-major and V transposed (per head) ----------------
__global__ void rope_pack(const float* __restrict__ qkv,
                          uint16_t* __restrict__ Qb, uint16_t* __restrict__ Kb,
                          uint16_t* __restrict__ Vt)
{
  int idx = blockIdx.x * blockDim.x + threadIdx.x;   // over B*S*H*D
  int d = idx & 63;
  int h = (idx >> 6) & 31;
  int s = (idx >> 11) & 2047;
  int b = idx >> 22;
  size_t rowbase = (size_t)(b * SS + s) * 3 * EE;
  size_t qoff = rowbase + (size_t)h * DD;
  size_t koff = rowbase + EE + (size_t)h * DD;
  size_t voff = rowbase + 2 * EE + (size_t)h * DD;

  float qv, kv;
  if (d < 32) {
    int j = d & 15;
    float inv = __powf(10000.0f, -(float)j / 16.0f);
    float fr = (float)s * inv;
    float sn, cs;
    __sincosf(fr, &sn, &cs);
    float q1 = qkv[qoff + j], q2 = qkv[qoff + j + 16];
    float k1 = qkv[koff + j], k2 = qkv[koff + j + 16];
    if (d < 16) { qv = q1 * cs - q2 * sn; kv = k1 * cs - k2 * sn; }
    else        { qv = q1 * sn + q2 * cs; kv = k1 * sn + k2 * cs; }
  } else {
    qv = qkv[qoff + d]; kv = qkv[koff + d];
  }
  float vv = qkv[voff + d];
  size_t bh = (size_t)b * HH + h;
  Qb[(bh * SS + s) * DD + d] = f2bf(qv);
  Kb[(bh * SS + s) * DD + d] = f2bf(kv);
  Vt[(bh * DD + d) * SS + s] = f2bf(vv);
}

// ---------------- stage 3: flash attention (online softmax, WMMA QK^T and PV) ----------------
__global__ __launch_bounds__(256) void flash_attn(
    const uint16_t* __restrict__ Qb, const uint16_t* __restrict__ Kb,
    const uint16_t* __restrict__ Vt, uint16_t* __restrict__ ctx)
{
  __shared__ uint16_t lds_p[8][16][32];   // per-wave P tile staging (C-layout -> A-layout)

  const int lane = threadIdx.x & 31;
  const int wave = threadIdx.x >> 5;
  const int g    = lane >> 4;
  const int ln   = lane & 15;
  const int bh   = blockIdx.y;            // b*H + h
  const int q0   = blockIdx.x * 128 + wave * 16;
  const int b    = bh >> 5;
  const int h    = bh & 31;

  const uint16_t* Qbase = Qb + (size_t)bh * SS * DD;
  const uint16_t* Kbase = Kb + (size_t)bh * SS * DD;
  const uint16_t* Vbase = Vt + (size_t)bh * DD * SS;

  // Q fragments for the two K-steps over head dim (kk = 0, 32)
  ABFrag qf[2];
  const uint16_t* qrow = Qbase + (size_t)(q0 + ln) * DD;
#pragma unroll
  for (int t = 0; t < 2; ++t) {
    qf[t].q[0] = *(const uint4*)(qrow + 32 * t + 8 * g);
    qf[t].q[1] = *(const uint4*)(qrow + 32 * t + 16 + 8 * g);
  }

  float m_row[8], l_row[8], cf[8];
  v8f o[4] = {};
#pragma unroll
  for (int r = 0; r < 8; ++r) { m_row[r] = -3.0e38f; l_row[r] = 0.0f; }
  const float scale = 0.125f;             // 1/sqrt(64)
  const int qmax = q0 + 15;

  for (int n0 = 0; n0 <= qmax; n0 += 32) {
    // ---- scores: two 16x16 tiles (keys n0..+15 and n0+16..+31) ----
    v8f sc[2] = {};
#pragma unroll
    for (int t2 = 0; t2 < 2; ++t2) {
      const uint16_t* krow = Kbase + (size_t)(n0 + 16 * t2 + ln) * DD;
#pragma unroll
      for (int t = 0; t < 2; ++t) {
        ABFrag kf;
        kf.q[0] = *(const uint4*)(krow + 32 * t + 8 * g);
        kf.q[1] = *(const uint4*)(krow + 32 * t + 16 + 8 * g);
        sc[t2] = __builtin_amdgcn_wmma_f32_16x16x32_bf16(false, qf[t].v, false, kf.v,
                                                         (short)0, sc[t2], false, false);
      }
    }

    // ---- mask + online softmax (row stats replicated across each 16-lane half) ----
    float p0[8], p1[8];
#pragma unroll
    for (int r = 0; r < 8; ++r) {
      int qr = q0 + r + 8 * g;
      float v0 = sc[0][r] * scale;
      float v1 = sc[1][r] * scale;
      v0 = (n0 + ln      > qr) ? NEG_MASK : v0;
      v1 = (n0 + 16 + ln > qr) ? NEG_MASK : v1;
      p0[r] = v0; p1[r] = v1;
      float tm = fmaxf(v0, v1);
#pragma unroll
      for (int off = 1; off < 16; off <<= 1)
        tm = fmaxf(tm, __shfl_xor(tm, off, 32));
      float nm = fmaxf(m_row[r], tm);
      cf[r] = __expf(m_row[r] - nm);
      m_row[r] = nm;
      p0[r] = __expf(p0[r] - nm);
      p1[r] = __expf(p1[r] - nm);
      float rs = p0[r] + p1[r];
#pragma unroll
      for (int off = 1; off < 16; off <<= 1)
        rs += __shfl_xor(rs, off, 32);
      l_row[r] = l_row[r] * cf[r] + rs;
      lds_p[wave][r + 8 * g][ln]      = f2bf(p0[r]);
      lds_p[wave][r + 8 * g][16 + ln] = f2bf(p1[r]);
    }
#pragma unroll
    for (int t = 0; t < 4; ++t)
#pragma unroll
      for (int r = 0; r < 8; ++r)
        o[t][r] = o[t][r] * cf[r];

    // intra-wave LDS RAW: lanes exchange P; DS ops are in-order per wave, wait for data
    asm volatile("s_wait_dscnt 0" ::: "memory");

    // reload P in A-matrix layout (row = ln, two contiguous 8-element halves)
    ABFrag pf;
    pf.q[0] = *(const uint4*)(&lds_p[wave][ln][8 * g]);
    pf.q[1] = *(const uint4*)(&lds_p[wave][ln][16 + 8 * g]);

    // ---- PV: accumulate into 4 d-tiles (V stored transposed: contiguous along keys) ----
#pragma unroll
    for (int t = 0; t < 4; ++t) {
      const uint16_t* vrow = Vbase + (size_t)(16 * t + ln) * SS + n0;
      ABFrag vf;
      vf.q[0] = *(const uint4*)(vrow + 8 * g);
      vf.q[1] = *(const uint4*)(vrow + 16 + 8 * g);
      o[t] = __builtin_amdgcn_wmma_f32_16x16x32_bf16(false, pf.v, false, vf.v,
                                                     (short)0, o[t], false, false);
    }
  }

  // ---- epilogue: normalize and write ctx[b, s, h*64+d] as bf16 ----
#pragma unroll
  for (int t = 0; t < 4; ++t) {
#pragma unroll
    for (int r = 0; r < 8; ++r) {
      int row = q0 + r + 8 * g;
      int col = h * DD + 16 * t + ln;
      ctx[(size_t)(b * SS + row) * EE + col] = f2bf(o[t][r] / l_row[r]);
    }
  }
}

// ---------------------------------- host launcher ----------------------------------
extern "C" void kernel_launch(void* const* d_in, const int* in_sizes, int n_in,
                              void* d_out, int out_size, void* d_ws, size_t ws_size,
                              hipStream_t stream) {
  const float* x    = (const float*)d_in[0];  // (2, 2048, 2048)
  const float* Wqkv = (const float*)d_in[1];  // (2048, 6144)
  const float* bqkv = (const float*)d_in[2];  // (6144,)
  const float* Wout = (const float*)d_in[3];  // (2048, 2048)
  const float* bout = (const float*)d_in[4];  // (2048,)
  float* out = (float*)d_out;

  char* ws = (char*)d_ws;
  const size_t M = (size_t)BB * SS;       // 4096
  uint16_t* xb     = (uint16_t*)(ws);                                  // 16 MB
  uint16_t* WqkvT  = (uint16_t*)(ws + 16777216);                       // 24 MB
  uint16_t* WoutT  = (uint16_t*)(ws + 41943040);                       // 8 MB
  float*    qkv    = (float*)   (ws + 50331648);                       // 96 MB
  uint16_t* Qb     = (uint16_t*)(ws + 150994944);                      // 16 MB
  uint16_t* Kb     = (uint16_t*)(ws + 167772160);                      // 16 MB
  uint16_t* Vt     = (uint16_t*)(ws + 184549376);                      // 16 MB
  uint16_t* ctxb   = (uint16_t*)(ws + 201326592);                      // 16 MB

  // stage 0: conversions
  {
    int n = (int)(M * EE);                                  // 8388608
    cvt_bf16<<<n / 256, 256, 0, stream>>>(x, xb, n);
    cvt_transpose_bf16<<<(EE * 3 * EE) / 256, 256, 0, stream>>>(Wqkv, WqkvT, EE, 3 * EE);
    cvt_transpose_bf16<<<(EE * EE) / 256, 256, 0, stream>>>(Wout, WoutT, EE, EE);
  }

  // stage 1: qkv = x @ Wqkv + bqkv   (M=4096, N=6144, K=2048)
  {
    dim3 grid(3 * EE / 128, (int)M / 128);
    gemm_bf16_lds<<<grid, 256, 0, stream>>>(xb, WqkvT, bqkv, qkv, (int)M, 3 * EE, EE);
  }

  // stage 2: RoPE + pack
  {
    int n = BB * SS * HH * DD;                              // 8388608
    rope_pack<<<n / 256, 256, 0, stream>>>(qkv, Qb, Kb, Vt);
  }

  // stage 3: causal flash attention -> ctxb (bf16, [b, s, e])
  {
    dim3 grid(SS / 128, BB * HH);
    flash_attn<<<grid, 256, 0, stream>>>(Qb, Kb, Vt, ctxb);
  }

  // stage 4: out = ctx @ Wout + bout  (M=4096, N=2048, K=2048)
  {
    dim3 grid(EE / 128, (int)M / 128);
    gemm_bf16_lds<<<grid, 256, 0, stream>>>(ctxb, WoutT, bout, out, (int)M, EE, EE);
  }
}
